// MambaLayer_44306882625891
// MI455X (gfx1250) — compile-verified, matrix operand order checked by reading, not run on hardware
//
#include <hip/hip_runtime.h>
#include <hip/hip_bf16.h>
#include <math.h>

// ---------------------------------------------------------------------------
// Mamba 3D layer for MI455X (gfx1250, wave32, WMMA, TDM).
// GEMMs: bf16 operands, f32 accumulate via v_wmma_f32_16x16x32_bf16, operands
// pre-swizzled to per-lane fragment layout; 1x4 N-blocking per wave where the
// N extent allows (A-fragment reuse x4).
// Scan: f32, 768 parallel channels, 16 states in registers; B/C tiles staged
// into LDS with the Tensor Data Mover (tensor_load_to_lds + s_wait_tensorcnt).
// ---------------------------------------------------------------------------

typedef __attribute__((ext_vector_type(16))) __bf16       v16bf;
typedef __attribute__((ext_vector_type(8)))  float        v8f;
typedef __attribute__((ext_vector_type(4)))  unsigned int u32x4;
typedef __attribute__((ext_vector_type(8)))  int          i32x8;
typedef __attribute__((ext_vector_type(4)))  int          i32x4;

#define BB   2
#define CC   192
#define LL   16384
#define BLR  (BB * LL)      // 32768 rows
#define DI   384            // d_inner
#define DSN  16             // d_state
#define MT   (BLR / 16)     // 2048 M-tiles

// ISA 16-bit A-matrix 16x32 per-lane k offsets (VGPR v, lane half):
//   v0: k=0,1 | 8,9   v1: 2,3 | 10,11  v2: 4,5 | 12,13  v3: 6,7 | 14,15
//   v4..7: +16
__device__ __forceinline__ int frag_k(int v, int lane) {
    return ((v & 3) << 1) + ((v >> 2) << 4) + ((lane >> 4) << 3);
}

// ---- pack weight W[N x K] (row major) into B-fragments (B = W^T, K x N) ----
__global__ void pack_B_kernel(const float* __restrict__ W, __bf16* __restrict__ dst,
                              int Nreal, int Ntiles, int Kreal, int Kb) {
    int idx = blockIdx.x * blockDim.x + threadIdx.x;
    if (idx >= Ntiles * Kb * 32) return;
    int lane = idx & 31;
    int kb   = (idx >> 5) % Kb;
    int nt   = (idx >> 5) / Kb;
    int n    = nt * 16 + (lane & 15);
    v16bf frag;
#pragma unroll
    for (int v = 0; v < 8; ++v) {
        int k = kb * 32 + frag_k(v, lane);
        float f0 = 0.f, f1 = 0.f;
        if (n < Nreal) {
            if (k     < Kreal) f0 = W[(size_t)n * Kreal + k];
            if (k + 1 < Kreal) f1 = W[(size_t)n * Kreal + k + 1];
        }
        frag[2 * v]     = (__bf16)f0;
        frag[2 * v + 1] = (__bf16)f1;
    }
    ((v16bf*)dst)[idx] = frag;
}

// ---- pack activation S[BLR x ld] rows into A-fragments -------------------
__global__ void pack_A_kernel(const float* __restrict__ S, __bf16* __restrict__ dst,
                              int ld, int Kreal, int Kb) {
    int idx = blockIdx.x * blockDim.x + threadIdx.x;
    if (idx >= MT * Kb * 32) return;
    int lane = idx & 31;
    int kb   = (idx >> 5) % Kb;
    int mt   = (idx >> 5) / Kb;
    int row  = mt * 16 + (lane & 15);
    v16bf frag;
#pragma unroll
    for (int v = 0; v < 8; ++v) {
        int k = kb * 32 + frag_k(v, lane);
        float f0 = (k     < Kreal) ? S[(size_t)row * ld + k]     : 0.f;
        float f1 = (k + 1 < Kreal) ? S[(size_t)row * ld + k + 1] : 0.f;
        frag[2 * v]     = (__bf16)f0;
        frag[2 * v + 1] = (__bf16)f1;
    }
    ((v16bf*)dst)[idx] = frag;
}

// ---- instance-norm statistics: one block per (b,c) ------------------------
__global__ void norm_stats_kernel(const float* __restrict__ x,
                                  float* __restrict__ mean, float* __restrict__ invstd) {
    int bc = blockIdx.x;                       // 0..383
    const float* p = x + (size_t)bc * LL;
    float s = 0.f, s2 = 0.f;
    for (int i = threadIdx.x; i < LL; i += blockDim.x) {
        float v = p[i]; s += v; s2 += v * v;
    }
    __shared__ float rs[256], rq[256];
    rs[threadIdx.x] = s; rq[threadIdx.x] = s2;
    __syncthreads();
    for (int st = 128; st > 0; st >>= 1) {
        if (threadIdx.x < st) {
            rs[threadIdx.x] += rs[threadIdx.x + st];
            rq[threadIdx.x] += rq[threadIdx.x + st];
        }
        __syncthreads();
    }
    if (threadIdx.x == 0) {
        float m   = rs[0] * (1.f / LL);
        float var = rq[0] * (1.f / LL) - m * m;
        mean[bc]   = m;
        invstd[bc] = rsqrtf(var + 1e-5f);
    }
}

// ---- normalize + transpose (B,C,L)->(BL,C) directly into A-fragments -----
__global__ void norm_pack_kernel(const float* __restrict__ x,
                                 const float* __restrict__ mean,
                                 const float* __restrict__ invstd,
                                 __bf16* __restrict__ dst) {
    const int Kb = CC / 32;                    // 6
    int idx = blockIdx.x * blockDim.x + threadIdx.x;
    if (idx >= MT * Kb * 32) return;
    int lane = idx & 31;
    int kb   = (idx >> 5) % Kb;
    int mt   = (idx >> 5) / Kb;
    int row  = mt * 16 + (lane & 15);
    int b    = row >> 14;                      // /LL
    int l    = row & (LL - 1);
    v16bf frag;
#pragma unroll
    for (int v = 0; v < 8; ++v) {
        int c0 = kb * 32 + frag_k(v, lane);
#pragma unroll
        for (int e = 0; e < 2; ++e) {
            int c = c0 + e;
            float val = (x[((size_t)(b * CC + c)) * LL + l] - mean[b * CC + c])
                        * invstd[b * CC + c];
            frag[2 * v + e] = (__bf16)val;
        }
    }
    ((v16bf*)dst)[idx] = frag;
}

// ---- WMMA GEMM, 1x1 tile per wave (for small Nt) -------------------------
__global__ void gemm_bf16_kernel(const __bf16* __restrict__ Af, const __bf16* __restrict__ Bf,
                                 float* __restrict__ Cmat, int Mtl, int Nt, int Kb, int ldc) {
    int wid  = (blockIdx.x * blockDim.x + threadIdx.x) >> 5;
    int lane = threadIdx.x & 31;
    if (wid >= Mtl * Nt) return;
    int mt = wid / Nt, nt = wid % Nt;
    const v16bf* Ap = (const v16bf*)Af + (size_t)mt * Kb * 32 + lane;
    const v16bf* Bp = (const v16bf*)Bf + (size_t)nt * Kb * 32 + lane;
    v8f acc = {0.f, 0.f, 0.f, 0.f, 0.f, 0.f, 0.f, 0.f};
    for (int kb = 0; kb < Kb; ++kb) {
        v16bf a = Ap[kb * 32];
        v16bf b = Bp[kb * 32];
        if (kb + 1 < Kb) __builtin_prefetch(Ap + (kb + 1) * 32, 0, 3);
        acc = __builtin_amdgcn_wmma_f32_16x16x32_bf16(
            false, a, false, b, (short)0, acc, false, false);
    }
    // D layout: lane -> N = lane&15, VGPR r -> M = r + 8*(lane>>4)
    int col = nt * 16 + (lane & 15);
    int r0  = mt * 16 + ((lane >> 4) << 3);
#pragma unroll
    for (int r = 0; r < 8; ++r)
        Cmat[(size_t)(r0 + r) * ldc + col] = acc[r];
}

// ---- WMMA GEMM, 1x4 N-blocking per wave (A-fragment reuse x4) ------------
__global__ void gemm_bf16_n4_kernel(const __bf16* __restrict__ Af, const __bf16* __restrict__ Bf,
                                    float* __restrict__ Cmat, int Mtl, int Ng, int Kb, int ldc) {
    int wid  = (blockIdx.x * blockDim.x + threadIdx.x) >> 5;
    int lane = threadIdx.x & 31;
    if (wid >= Mtl * Ng) return;
    int mt = wid / Ng, ng = wid % Ng;
    const v16bf* Ap = (const v16bf*)Af + (size_t)mt * Kb * 32 + lane;
    const v16bf* Bp = (const v16bf*)Bf + (size_t)(ng * 4) * Kb * 32 + lane;
    v8f acc[4];
#pragma unroll
    for (int j = 0; j < 4; ++j) acc[j] = (v8f){0.f, 0.f, 0.f, 0.f, 0.f, 0.f, 0.f, 0.f};
    for (int kb = 0; kb < Kb; ++kb) {
        v16bf a = Ap[kb * 32];
        if (kb + 1 < Kb) __builtin_prefetch(Ap + (kb + 1) * 32, 0, 3);
        v16bf bfr[4];
#pragma unroll
        for (int j = 0; j < 4; ++j) bfr[j] = Bp[(size_t)j * Kb * 32 + kb * 32];
#pragma unroll
        for (int j = 0; j < 4; ++j)
            acc[j] = __builtin_amdgcn_wmma_f32_16x16x32_bf16(
                false, a, false, bfr[j], (short)0, acc[j], false, false);
    }
    int r0 = mt * 16 + ((lane >> 4) << 3);
#pragma unroll
    for (int j = 0; j < 4; ++j) {
        int col = (ng * 4 + j) * 16 + (lane & 15);
#pragma unroll
        for (int r = 0; r < 8; ++r)
            Cmat[(size_t)(r0 + r) * ldc + col] = acc[j][r];
    }
}

// ---- final GEMM: fuse bias add + (BL,C)->(B,C,L) transposed store --------
__global__ void gemm_final_kernel(const __bf16* __restrict__ Af, const __bf16* __restrict__ Bf,
                                  const float* __restrict__ bias, float* __restrict__ out,
                                  int Mtl, int Nt, int Kb) {
    int wid  = (blockIdx.x * blockDim.x + threadIdx.x) >> 5;
    int lane = threadIdx.x & 31;
    if (wid >= Mtl * Nt) return;
    int mt = wid / Nt, nt = wid % Nt;
    const v16bf* Ap = (const v16bf*)Af + (size_t)mt * Kb * 32 + lane;
    const v16bf* Bp = (const v16bf*)Bf + (size_t)nt * Kb * 32 + lane;
    v8f acc = {0.f, 0.f, 0.f, 0.f, 0.f, 0.f, 0.f, 0.f};
    for (int kb = 0; kb < Kb; ++kb) {
        v16bf a = Ap[kb * 32];
        v16bf b = Bp[kb * 32];
        acc = __builtin_amdgcn_wmma_f32_16x16x32_bf16(
            false, a, false, b, (short)0, acc, false, false);
    }
    int col = nt * 16 + (lane & 15);
    float bv = bias[col];
    int r0 = ((lane >> 4) << 3);
#pragma unroll
    for (int r = 0; r < 8; ++r) {
        int row = mt * 16 + r0 + r;            // row = b*LL + l
        int b   = row >> 14;
        int l   = row & (LL - 1);
        out[((size_t)(b * CC + col)) * LL + l] = acc[r] + bv;   // contiguous in l
    }
}

// ---- depthwise causal conv (K=4) + SiLU ----------------------------------
__global__ void conv_silu_kernel(const float* __restrict__ xz, const float* __restrict__ cw,
                                 const float* __restrict__ cb, float* __restrict__ uc) {
    int idx = blockIdx.x * blockDim.x + threadIdx.x;
    if (idx >= BLR * DI) return;
    int d  = idx % DI;
    int bl = idx / DI;
    int l  = bl & (LL - 1);
    float acc = cb[d];
#pragma unroll
    for (int k = 0; k < 4; ++k) {
        int lk = l - 3 + k;
        if (lk >= 0) acc += xz[(size_t)(bl - 3 + k) * (2 * DI) + d] * cw[d * 4 + k];
    }
    uc[idx] = acc / (1.f + __expf(-acc));      // silu
}

// ---- dt = softplus(dt_pre + bias), in place ------------------------------
__global__ void softplus_kernel(float* __restrict__ dt, const float* __restrict__ bias) {
    int idx = blockIdx.x * blockDim.x + threadIdx.x;
    if (idx >= BLR * DI) return;
    float v = dt[idx] + bias[idx % DI];
    dt[idx] = (v > 20.f) ? v : log1pf(__expf(v));
}

// ---- A = -exp(A_log) -----------------------------------------------------
__global__ void prep_A_kernel(const float* __restrict__ Al, float* __restrict__ An) {
    int idx = blockIdx.x * blockDim.x + threadIdx.x;
    if (idx < DI * DSN) An[idx] = -__expf(Al[idx]);
}

// ---- selective scan: 1 thread per (b,d), 16 states in regs ---------------
// B/C rows (64 steps x 32 floats, global row stride 48 floats) are staged in
// LDS by the Tensor Data Mover; one TDM descriptor per chunk, issued by the
// first wave (uniform branch: TDM ignores EXEC), completion via TENSORcnt.
__global__ void scan_kernel(const float* __restrict__ uc, const float* __restrict__ dt,
                            const float* __restrict__ dbl, const float* __restrict__ An,
                            float* __restrict__ ys) {
    __shared__ float sBC[64 * 32];             // 64 steps x (16 B + 16 C)
    int tid = threadIdx.x;                     // 0..63
    int grp = blockIdx.x;                      // 0..11
    int b   = grp / 6;
    int d   = (grp % 6) * 64 + tid;

    // LDS byte offset of sBC (addrspacecast generic -> AS3, truncate)
    unsigned lds_base =
        (unsigned)(unsigned long long)(__attribute__((address_space(3))) char*)&sBC[0];

    // D# group1 (constant): data_size=4B (code 2), tensor dims huge (no OOB),
    // tile 32 x 64, tensor_dim0_stride = 48 elements.
    //   q0 = data_size<<16                                  -> w0,w1
    //   q1 = td0[31:16] | td1<<16 | tile_dim0<<48           -> w2,w3
    //   q2 = tile_dim1 | stride0<<32                        -> w4,w5
    const i32x8 g1  = {0x20000, 0, 0x4000, 0x204000, 64, 48, 0, 0};
    const i32x4 gz4 = {0, 0, 0, 0};
    const i32x8 gz8 = {0, 0, 0, 0, 0, 0, 0, 0};

    float a[DSN], h[DSN];
#pragma unroll
    for (int s = 0; s < DSN; ++s) { a[s] = An[d * DSN + s]; h[s] = 0.f; }

    unsigned long long gbase = (unsigned long long)(size_t)dbl;
    for (int l0 = 0; l0 < LL; l0 += 64) {
        __syncthreads();                       // previous chunk fully consumed
        if (tid < 32) {                        // wave 0 issues the TDM load
            unsigned long long ga =
                gbase + ((unsigned long long)(b * LL + l0) * 48 + 12) * 4;
            u32x4 g0;
            g0[0] = 1u;                                        // count=1
            g0[1] = lds_base;                                  // lds_addr
            g0[2] = (unsigned)ga;                              // global_addr lo
            g0[3] = (unsigned)((ga >> 32) & 0x1FFFFFFu) | (2u << 30);  // hi | type=2
            __builtin_amdgcn_tensor_load_to_lds(g0, g1, gz4, gz4, gz8, 0);
            __builtin_amdgcn_s_wait_tensorcnt(0);
        }
        __syncthreads();                       // tile visible to both waves
        for (int t = 0; t < 64; ++t) {
            int bl = b * LL + l0 + t;
            float dv = dt[(size_t)bl * DI + d];
            float uv = uc[(size_t)bl * DI + d];
            float du = dv * uv;
            const float* Bv = &sBC[t * 32];
            float y = 0.f;
#pragma unroll
            for (int s = 0; s < DSN; ++s) {
                float dA = __expf(dv * a[s]);
                h[s] = dA * h[s] + du * Bv[s];
                y += h[s] * Bv[16 + s];
            }
            ys[(size_t)bl * DI + d] = y;
        }
    }
}

// ---- y = (ys + uc*D) * silu(z), in place into ys -------------------------
__global__ void finalize_kernel(float* __restrict__ ys, const float* __restrict__ uc,
                                const float* __restrict__ xz, const float* __restrict__ Dp) {
    int idx = blockIdx.x * blockDim.x + threadIdx.x;
    if (idx >= BLR * DI) return;
    int d  = idx % DI;
    int bl = idx / DI;
    float z = xz[(size_t)bl * (2 * DI) + DI + d];
    float y = ys[idx] + uc[idx] * Dp[d];
    ys[idx] = y * (z / (1.f + __expf(-z)));
}

// ---------------------------------------------------------------------------
extern "C" void kernel_launch(void* const* d_in, const int* in_sizes, int n_in,
                              void* d_out, int out_size, void* d_ws, size_t ws_size,
                              hipStream_t stream) {
    (void)in_sizes; (void)n_in; (void)out_size; (void)ws_size;
    const float* x      = (const float*)d_in[0];
    const float* w_in   = (const float*)d_in[1];
    const float* conv_w = (const float*)d_in[2];
    const float* conv_b = (const float*)d_in[3];
    const float* w_x    = (const float*)d_in[4];
    const float* w_dt   = (const float*)d_in[5];
    const float* b_dt   = (const float*)d_in[6];
    const float* A_log  = (const float*)d_in[7];
    const float* Dp     = (const float*)d_in[8];
    const float* w_out  = (const float*)d_in[9];
    const float* w_pr   = (const float*)d_in[10];
    const float* b_pr   = (const float*)d_in[11];
    float* out = (float*)d_out;

    char* wsb = (char*)d_ws;
    size_t off = 0;
    auto alloc = [&](size_t bytes) -> char* {
        char* p = wsb + off;
        off = (off + bytes + 255) & ~(size_t)255;
        return p;
    };
    // fragment buffers: tiles * 32 lanes * 16 bf16 * 2B = tiles * 1024 B
    __bf16* wbin  = (__bf16*)alloc((size_t)48 * 6  * 1024);  // in_proj  B-frags
    __bf16* wbx   = (__bf16*)alloc((size_t)3  * 12 * 1024);  // x_proj   B-frags (N pad 44->48)
    __bf16* wbdt  = (__bf16*)alloc((size_t)24 * 1  * 1024);  // dt_proj  B-frags (K pad 12->32)
    __bf16* wbout = (__bf16*)alloc((size_t)12 * 12 * 1024);  // out_proj B-frags
    __bf16* wbpr  = (__bf16*)alloc((size_t)12 * 6  * 1024);  // proj     B-frags
    float*  An    = (float*) alloc((size_t)DI * DSN * 4);
    float*  meanb = (float*) alloc((size_t)BB * CC * 4);
    float*  istdb = (float*) alloc((size_t)BB * CC * 4);
    __bf16* xfA   = (__bf16*)alloc((size_t)MT * 6  * 1024);  // normalized input A-frags
    float*  xz    = (float*) alloc((size_t)BLR * 768 * 4);   // [u | z]
    float*  uc    = (float*) alloc((size_t)BLR * DI * 4);    // conv+silu output
    __bf16* ucA   = (__bf16*)alloc((size_t)MT * 12 * 1024);
    float*  dbl   = (float*) alloc((size_t)BLR * 48 * 4);    // [dt_low(12)|B(16)|C(16)|pad]
    __bf16* dtA   = (__bf16*)alloc((size_t)MT * 1  * 1024);
    float*  dtb   = (float*) alloc((size_t)BLR * DI * 4);    // dt pre/post softplus
    float*  ysb   = (float*) alloc((size_t)BLR * DI * 4);    // scan out -> fused y
    __bf16* yA    = (__bf16*)alloc((size_t)MT * 12 * 1024);
    float*  mo    = (float*) alloc((size_t)BLR * CC * 4);    // out_proj result
    __bf16* moA   = (__bf16*)alloc((size_t)MT * 6  * 1024);

    // --- weight prep (small) ---
    pack_B_kernel<<<(48 * 6 * 32 + 255) / 256, 256, 0, stream>>>(w_in,  wbin,  768, 48, 192, 6);
    pack_B_kernel<<<(3 * 12 * 32 + 255) / 256, 256, 0, stream>>>(w_x,   wbx,   44,  3,  384, 12);
    pack_B_kernel<<<(24 * 1 * 32 + 255) / 256, 256, 0, stream>>>(w_dt,  wbdt,  384, 24, 12,  1);
    pack_B_kernel<<<(12 * 12 * 32 + 255) / 256, 256, 0, stream>>>(w_out, wbout, 192, 12, 384, 12);
    pack_B_kernel<<<(12 * 6 * 32 + 255) / 256, 256, 0, stream>>>(w_pr,  wbpr,  192, 12, 192, 6);
    prep_A_kernel<<<(DI * DSN + 255) / 256, 256, 0, stream>>>(A_log, An);

    // --- instance norm + transpose into A-fragments ---
    norm_stats_kernel<<<BB * CC, 256, 0, stream>>>(x, meanb, istdb);
    norm_pack_kernel<<<MT * 6 * 32 / 256, 256, 0, stream>>>(x, meanb, istdb, xfA);

    // --- in_proj: (BL x 192) @ (192 x 768), 1x4 blocking ---
    gemm_bf16_n4_kernel<<<MT * 12 / 8, 256, 0, stream>>>(xfA, wbin, xz, MT, 12, 6, 768);

    // --- depthwise conv + silu ---
    conv_silu_kernel<<<BLR * DI / 256, 256, 0, stream>>>(xz, conv_w, conv_b, uc);
    pack_A_kernel<<<MT * 12 * 32 / 256, 256, 0, stream>>>(uc, ucA, DI, DI, 12);

    // --- x_proj: (BL x 384) @ (384 x 44->48) ---
    gemm_bf16_kernel<<<MT * 3 / 8, 256, 0, stream>>>(ucA, wbx, dbl, MT, 3, 12, 48);

    // --- dt_proj: (BL x 12) @ (12 x 384), then softplus(+bias) ---
    pack_A_kernel<<<MT * 1 * 32 / 256, 256, 0, stream>>>(dbl, dtA, 48, 12, 1);
    gemm_bf16_n4_kernel<<<MT * 6 / 8, 256, 0, stream>>>(dtA, wbdt, dtb, MT, 6, 1, DI);
    softplus_kernel<<<BLR * DI / 256, 256, 0, stream>>>(dtb, b_dt);

    // --- selective scan (sequential over L, parallel over b*d_inner) ---
    scan_kernel<<<BB * 6, 64, 0, stream>>>(uc, dtb, dbl, An, ysb);

    // --- gate: y = (ys + uc*D) * silu(z) ---
    finalize_kernel<<<BLR * DI / 256, 256, 0, stream>>>(ysb, uc, xz, Dp);
    pack_A_kernel<<<MT * 12 * 32 / 256, 256, 0, stream>>>(ysb, yA, DI, DI, 12);

    // --- out_proj: (BL x 384) @ (384 x 192), 1x4 blocking ---
    gemm_bf16_n4_kernel<<<MT * 3 / 8, 256, 0, stream>>>(yA, wbout, mo, MT, 3, 12, CC);

    // --- final proj + bias + transposed store to (B,C,D,H,W) ---
    pack_A_kernel<<<MT * 6 * 32 / 256, 256, 0, stream>>>(mo, moA, CC, CC, 6);
    gemm_final_kernel<<<MT * 12 / 8, 256, 0, stream>>>(moA, wbpr, b_pr, out, MT, 12, 6);
}
